// Agent_45535243272416
// MI455X (gfx1250) — matrix operand order repaired and checked by reading, not compiled, not run on hardware
//
#include <hip/hip_runtime.h>
#include <hip/hip_bf16.h>

// ---------------- problem constants ----------------
#define B_SZ    16
#define HIDDEN  1024
#define MEMLEN  512
#define L_SEQ   513          // MEMLEN + 1
#define D_STATE 16
#define D_CONV  4
#define D_INNER 2048
#define DT_RANK 64
#define N_ACT   18
#define BL      (B_SZ * L_SEQ)   // 8208 tokens

typedef __bf16 bf16;
typedef __attribute__((ext_vector_type(16))) __bf16 v16bf;
typedef __attribute__((ext_vector_type(8)))  float  v8f;

// ---------------- device helpers ----------------
__device__ __forceinline__ float sigmoidf_fast(float x) {
    return 1.0f / (1.0f + __expf(-x));
}
__device__ __forceinline__ float siluf(float x) { return x * sigmoidf_fast(x); }
__device__ __forceinline__ float softplusf(float x) {
    return (x > 20.0f) ? x : __logf(1.0f + __expf(x));
}

// ---- WMMA fragment loads (wave32; contiguous k = 16*(lane/16) + s, same order in A and B) ----
// A (16x32 bf16, row major): lane L (0..15) holds row M=L, k = 0..15;
// lanes 16..31 hold row M=L-16, k = 16..31.
// Ragged M handled by CLAMPING the row (garbage lands only in accumulator rows
// the store never writes), keeping the k-loop branch-free.
__device__ __forceinline__ v16bf load_a_frag(const bf16* __restrict__ A, long long lda,
                                             int m0, int k0, int M) {
    int lane = threadIdx.x & 31;
    int row  = m0 + (lane & 15);
    row = (row < M) ? row : (M - 1);          // clamp: dead rows read valid memory
    int kb   = (lane >> 4) * 16;
    const bf16* p = A + (long long)row * lda + (k0 + kb);
    v16bf a;
#pragma unroll
    for (int j = 0; j < 8; ++j) a[j] = p[j];
#pragma unroll
    for (int j = 0; j < 8; ++j) a[8 + j] = p[8 + j];
    return a;
}

// B (32x16 bf16). We compute X @ W^T, so B(k,n) = W[n, k].  Lanes 0..15 carry
// n=lane with k=0..15; lanes 16..31 carry n=lane-16 with k=16..31 (contiguous
// 16 bf16 per lane -> two b128 loads from row-major W). Ragged N clamped.
__device__ __forceinline__ v16bf load_b_frag(const bf16* __restrict__ W, long long ldw,
                                             int n0, int k0, int N) {
    int lane = threadIdx.x & 31;
    int n    = n0 + (lane & 15);
    n = (n < N) ? n : (N - 1);                // clamp: dead cols read valid memory
    int kh   = (lane >> 4) * 16;
    const bf16* p = W + (long long)n * ldw + (k0 + kh);
    v16bf b;
#pragma unroll
    for (int j = 0; j < 16; ++j) b[j] = p[j];
    return b;
}

// C/D (16x16 f32): lane n = lane&15, VGPR r holds M = r + 8*(lane>>4).
// Fast path for full 16-row subtiles (the common case here: all M are
// multiples of 16): single base address, unguarded stores, no exec churn.
template <int OUT_BF16>
__device__ __forceinline__ void store_d_frag(void* __restrict__ Cout, long long ldc,
                                             int m0, int n0, v8f c, int mvalid) {
    int lane = threadIdx.x & 31;
    int n    = lane & 15;
    int mo   = (lane >> 4) * 8;
    long long base = (long long)(m0 + mo) * ldc + (n0 + n);
    if (mvalid == 16) {
#pragma unroll
        for (int vr = 0; vr < 8; ++vr) {
            if (OUT_BF16) ((bf16*)Cout)[base + (long long)vr * ldc] = (bf16)c[vr];
            else          ((float*)Cout)[base + (long long)vr * ldc] = c[vr];
        }
    } else {
#pragma unroll
        for (int vr = 0; vr < 8; ++vr) {
            if (mo + vr < mvalid) {
                if (OUT_BF16) ((bf16*)Cout)[base + (long long)vr * ldc] = (bf16)c[vr];
                else          ((float*)Cout)[base + (long long)vr * ldc] = c[vr];
            }
        }
    }
}

// ---------------- WMMA GEMM: C[M,N] = A[M,K] (bf16) @ W[N,K]^T (bf16) ----------------
// block = 256 threads = 8 waves arranged 2(M) x 4(N); wave tile = 32x64
// (8 WMMAs per 6 fragment loads). Block tile = 64x256.
// Requires K % 32 == 0, N % 16 == 0. M ragged (per-16-row store guard,
// clamped loads). No barriers -> divergent early exit is safe.
template <int OUT_BF16>
__global__ void wmma_gemm_kernel(const bf16* __restrict__ A, long long lda,
                                 const bf16* __restrict__ W, long long ldw,
                                 void* __restrict__ Cout, long long ldc,
                                 int M, int N, int K) {
    int wave = threadIdx.x >> 5;
    int wm = wave & 1;
    int wn = wave >> 1;
    int m0 = blockIdx.x * 64 + wm * 32;
    int n0 = blockIdx.y * 256 + wn * 64;
    if (m0 >= M || n0 >= N) return;
    int mv0 = M - m0;       if (mv0 > 16) mv0 = 16;
    int mv1 = M - m0 - 16;  if (mv1 > 16) mv1 = 16;  if (mv1 < 0) mv1 = 0;

    v8f acc0[4]; v8f acc1[4];
#pragma unroll
    for (int j = 0; j < 4; ++j) { acc0[j] = (v8f){}; acc1[j] = (v8f){}; }

    for (int k = 0; k < K; k += 32) {
        v16bf a0 = load_a_frag(A, lda, m0,      k, M);
        v16bf a1 = load_a_frag(A, lda, m0 + 16, k, M);
        v16bf b[4];
#pragma unroll
        for (int j = 0; j < 4; ++j) b[j] = load_b_frag(W, ldw, n0 + 16 * j, k, N);
#pragma unroll
        for (int j = 0; j < 4; ++j)
            acc0[j] = __builtin_amdgcn_wmma_f32_16x16x32_bf16(false, a0, false, b[j],
                                                              (short)0, acc0[j], false, false);
#pragma unroll
        for (int j = 0; j < 4; ++j)
            acc1[j] = __builtin_amdgcn_wmma_f32_16x16x32_bf16(false, a1, false, b[j],
                                                              (short)0, acc1[j], false, false);
    }
#pragma unroll
    for (int j = 0; j < 4; ++j) {
        int nj = n0 + 16 * j;
        if (nj < N) {                       // N is a multiple of 16: all-or-nothing
            store_d_frag<OUT_BF16>(Cout, ldc, m0,      nj, acc0[j], mv0);
            store_d_frag<OUT_BF16>(Cout, ldc, m0 + 16, nj, acc1[j], mv1);
        }
    }
}

// ---------------- elementwise kernels ----------------
__global__ void cvt_f32_bf16_kernel(const float* __restrict__ src, bf16* __restrict__ dst,
                                    long long n) {
    long long i = (long long)blockIdx.x * blockDim.x + threadIdx.x;
    if (i < n) dst[i] = (bf16)src[i];
}

// build seqB[b, l, h] (bf16): l < 512 -> memory_window; l == 512 -> relu(x @ enc_W^T + enc_b)
__global__ void build_seq_kernel(const float* __restrict__ mem, const float* __restrict__ x,
                                 const float* __restrict__ enc_W, const float* __restrict__ enc_b,
                                 bf16* __restrict__ seqB) {
    long long idx = (long long)blockIdx.x * blockDim.x + threadIdx.x;
    if (idx >= (long long)BL * HIDDEN) return;
    int h  = (int)(idx % HIDDEN);
    int bl = (int)(idx / HIDDEN);
    int l  = bl % L_SEQ;
    int b  = bl / L_SEQ;
    float v;
    if (l < MEMLEN) {
        v = mem[((long long)b * MEMLEN + l) * HIDDEN + h];
    } else {
        v = enc_b[h];
        v += x[b * 3 + 0] * enc_W[h * 3 + 0];
        v += x[b * 3 + 1] * enc_W[h * 3 + 1];
        v += x[b * 3 + 2] * enc_W[h * 3 + 2];
        v = fmaxf(v, 0.0f);
    }
    seqB[idx] = (bf16)v;
}

// depthwise causal conv(D_CONV=4) + bias + SiLU.  xin layout [(b*L+l)*D_INNER + d].
__global__ void conv_silu_kernel(const bf16* __restrict__ xin, const float* __restrict__ convW,
                                 const float* __restrict__ convb, bf16* __restrict__ out) {
    long long idx = (long long)blockIdx.x * blockDim.x + threadIdx.x;
    if (idx >= (long long)BL * D_INNER) return;
    int d  = (int)(idx % D_INNER);
    int bl = (int)(idx / D_INNER);
    int l  = bl % L_SEQ;
    int b  = bl / L_SEQ;
    float acc = convb[d];
#pragma unroll
    for (int t = 0; t < D_CONV; ++t) {
        int ls = l - (D_CONV - 1) + t;
        if (ls >= 0)
            acc += convW[d * D_CONV + t] *
                   (float)xin[((long long)b * L_SEQ + ls) * D_INNER + d];
    }
    out[idx] = (bf16)siluf(acc);
}

// extract dt-rank slice of x_dbl into contiguous bf16 (for dt_proj GEMM A-matrix)
__global__ void extract_dt_kernel(const float* __restrict__ xdbl, bf16* __restrict__ dtB) {
    long long idx = (long long)blockIdx.x * blockDim.x + threadIdx.x;
    if (idx >= (long long)BL * DT_RANK) return;
    int c = (int)(idx % DT_RANK);
    int r = (int)(idx / DT_RANK);
    dtB[idx] = (bf16)xdbl[(long long)r * (DT_RANK + 2 * D_STATE) + c];
}

// selective scan: one thread per (b, d) channel, 16-state recurrence over L=513 steps.
// Only the final y (last timestep) is needed downstream.
#define SCAN_CHUNK 64
__global__ void scan_kernel(const float* __restrict__ dtraw, const float* __restrict__ dtb,
                            const bf16* __restrict__ usilu, const float* __restrict__ xdbl,
                            const float* __restrict__ A_log, const float* __restrict__ Dp,
                            const float* __restrict__ zlast,
                            float* __restrict__ ylast, bf16* __restrict__ ylastB) {
    const int XDW = DT_RANK + 2 * D_STATE;   // 96
    int tid = threadIdx.x;                   // 256 threads
    int b   = blockIdx.x >> 3;               // 16 batches
    int d   = ((blockIdx.x & 7) << 8) | tid; // 2048 channels

    float a[D_STATE];
#pragma unroll
    for (int n = 0; n < D_STATE; ++n) a[n] = -__expf(A_log[d * D_STATE + n]);
    float h[D_STATE];
#pragma unroll
    for (int n = 0; n < D_STATE; ++n) h[n] = 0.0f;

    __shared__ float Bsh[SCAN_CHUNK][D_STATE];
    const float dbias = dtb[d];
    float ulast = 0.0f;

    for (int l0 = 0; l0 < L_SEQ; l0 += SCAN_CHUNK) {
        int cnt = L_SEQ - l0; if (cnt > SCAN_CHUNK) cnt = SCAN_CHUNK;
        __syncthreads();
        for (int i = tid; i < cnt * D_STATE; i += 256) {
            int ll = i >> 4, n = i & 15;
            Bsh[ll][n] = xdbl[((long long)b * L_SEQ + l0 + ll) * XDW + DT_RANK + n];
        }
        __syncthreads();
        for (int li = 0; li < cnt; ++li) {
            long long row = (long long)b * L_SEQ + l0 + li;
            float dt = softplusf(dtraw[row * D_INNER + d] + dbias);
            float u  = (float)usilu[row * D_INNER + d];
            ulast = u;
            float dtu = dt * u;
#pragma unroll
            for (int n = 0; n < D_STATE; ++n)
                h[n] = __expf(dt * a[n]) * h[n] + dtu * Bsh[li][n];
        }
    }

    // y = h . C_last + u_last * D, gated by silu(z_last)
    long long lastrow = (long long)b * L_SEQ + (L_SEQ - 1);
    float y = 0.0f;
#pragma unroll
    for (int n = 0; n < D_STATE; ++n)
        y += h[n] * xdbl[lastrow * XDW + DT_RANK + D_STATE + n];
    y += ulast * Dp[d];
    y *= siluf(zlast[(long long)b * D_INNER + d]);
    ylast[(long long)b * D_INNER + d]  = y;
    ylastB[(long long)b * D_INNER + d] = (bf16)y;
}

// actor/critic heads: 16*18 logits + 16 values, 1024-length f32 dots.
__global__ void heads_kernel(const float* __restrict__ cur,
                             const float* __restrict__ actor_W, const float* __restrict__ actor_b,
                             const float* __restrict__ critic_W, const float* __restrict__ critic_b,
                             float* __restrict__ logits, float* __restrict__ value) {
    int i = threadIdx.x;
    if (i < B_SZ * N_ACT) {
        int b = i / N_ACT, a2 = i % N_ACT;
        float acc = actor_b[a2];
        for (int k = 0; k < HIDDEN; ++k)
            acc += cur[b * HIDDEN + k] * actor_W[a2 * HIDDEN + k];
        logits[i] = acc;
    } else if (i < B_SZ * N_ACT + B_SZ) {
        int b = i - B_SZ * N_ACT;
        float acc = critic_b[0];
        for (int k = 0; k < HIDDEN; ++k)
            acc += cur[b * HIDDEN + k] * critic_W[k];
        value[b] = acc;
    }
}

// new_memory[b, l] = memory[b, l+1] for l<511; new_memory[b, 511] = current[b]
__global__ void shift_mem_kernel(const float* __restrict__ mem, const float* __restrict__ cur,
                                 float* __restrict__ newmem) {
    long long idx = (long long)blockIdx.x * blockDim.x + threadIdx.x;
    if (idx >= (long long)B_SZ * MEMLEN * HIDDEN) return;
    int h  = (int)(idx % HIDDEN);
    int bl = (int)(idx / HIDDEN);
    int l  = bl % MEMLEN;
    int b  = bl / MEMLEN;
    float v = (l < MEMLEN - 1)
                  ? mem[((long long)b * MEMLEN + l + 1) * HIDDEN + h]
                  : cur[b * HIDDEN + h];
    newmem[idx] = v;
}

// ---------------- launch ----------------
extern "C" void kernel_launch(void* const* d_in, const int* in_sizes, int n_in,
                              void* d_out, int out_size, void* d_ws, size_t ws_size,
                              hipStream_t stream) {
    (void)in_sizes; (void)n_in; (void)out_size; (void)ws_size;

    const float* x          = (const float*)d_in[0];
    const float* mem        = (const float*)d_in[1];
    const float* enc_W      = (const float*)d_in[2];
    const float* enc_b      = (const float*)d_in[3];
    const float* in_proj_W  = (const float*)d_in[4];
    const float* conv_W     = (const float*)d_in[5];
    const float* conv_b     = (const float*)d_in[6];
    const float* x_proj_W   = (const float*)d_in[7];
    const float* dt_proj_W  = (const float*)d_in[8];
    const float* dt_proj_b  = (const float*)d_in[9];
    const float* A_log      = (const float*)d_in[10];
    const float* D_param    = (const float*)d_in[11];
    const float* out_proj_W = (const float*)d_in[12];
    const float* actor_W    = (const float*)d_in[13];
    const float* actor_b    = (const float*)d_in[14];
    const float* critic_W   = (const float*)d_in[15];
    const float* critic_b   = (const float*)d_in[16];

    // d_out layout: logits(16*18) | value(16) | new_memory(16*512*1024) | current(16*1024)
    float* out    = (float*)d_out;
    float* logits = out;
    float* value  = out + B_SZ * N_ACT;
    float* newmem = value + B_SZ;
    float* current = newmem + (size_t)B_SZ * MEMLEN * HIDDEN;

    // workspace carve-out (~169 MB)
    char* ws = (char*)d_ws;
    size_t off = 0;
    auto carve = [&](size_t bytes) -> void* {
        void* p = ws + off;
        off += (bytes + 255) & ~(size_t)255;
        return p;
    };
    bf16*  seqB    = (bf16*)carve((size_t)BL * HIDDEN * sizeof(bf16));             // 16.8 MB
    bf16*  WinB    = (bf16*)carve((size_t)2 * D_INNER * HIDDEN * sizeof(bf16));    //  8.4 MB
    bf16*  WxpB    = (bf16*)carve((size_t)(DT_RANK + 2 * D_STATE) * D_INNER * sizeof(bf16));
    bf16*  WdtB    = (bf16*)carve((size_t)D_INNER * DT_RANK * sizeof(bf16));
    bf16*  WoutB   = (bf16*)carve((size_t)HIDDEN * D_INNER * sizeof(bf16));        //  4.2 MB
    bf16*  xinRaw  = (bf16*)carve((size_t)BL * D_INNER * sizeof(bf16));            // 33.6 MB
    bf16*  xinSilu = (bf16*)carve((size_t)BL * D_INNER * sizeof(bf16));            // 33.6 MB
    float* xdbl    = (float*)carve((size_t)BL * (DT_RANK + 2 * D_STATE) * sizeof(float));
    bf16*  dtB     = (bf16*)carve((size_t)BL * DT_RANK * sizeof(bf16));
    float* dtraw   = (float*)carve((size_t)BL * D_INNER * sizeof(float));          // 67.2 MB
    float* zlast   = (float*)carve((size_t)B_SZ * D_INNER * sizeof(float));
    float* ylast   = (float*)carve((size_t)B_SZ * D_INNER * sizeof(float));
    bf16*  ylastB  = (bf16*)carve((size_t)B_SZ * D_INNER * sizeof(bf16));

    dim3 blk(256);
    auto cdiv = [](long long a, long long b) { return (unsigned)((a + b - 1) / b); };

    // 1) bf16 weight conversions
    cvt_f32_bf16_kernel<<<cdiv((long long)2 * D_INNER * HIDDEN, 256), blk, 0, stream>>>(in_proj_W, WinB, (long long)2 * D_INNER * HIDDEN);
    cvt_f32_bf16_kernel<<<cdiv((long long)(DT_RANK + 2 * D_STATE) * D_INNER, 256), blk, 0, stream>>>(x_proj_W, WxpB, (long long)(DT_RANK + 2 * D_STATE) * D_INNER);
    cvt_f32_bf16_kernel<<<cdiv((long long)D_INNER * DT_RANK, 256), blk, 0, stream>>>(dt_proj_W, WdtB, (long long)D_INNER * DT_RANK);
    cvt_f32_bf16_kernel<<<cdiv((long long)HIDDEN * D_INNER, 256), blk, 0, stream>>>(out_proj_W, WoutB, (long long)HIDDEN * D_INNER);

    // 2) encode + build bf16 sequence
    build_seq_kernel<<<cdiv((long long)BL * HIDDEN, 256), blk, 0, stream>>>(mem, x, enc_W, enc_b, seqB);

    // 3) GEMM1: xin = seq @ in_proj_W[0:2048]^T  (bf16 out, M=8208,N=2048,K=1024)
    wmma_gemm_kernel<1><<<dim3(cdiv(BL, 64), D_INNER / 256), blk, 0, stream>>>(
        seqB, HIDDEN, WinB, HIDDEN, xinRaw, D_INNER, BL, D_INNER, HIDDEN);

    // 3b) z at last token only: M=16 strided-row A (rows b*513+512), W rows 2048..4095
    wmma_gemm_kernel<0><<<dim3(1, D_INNER / 256), blk, 0, stream>>>(
        seqB + (size_t)MEMLEN * HIDDEN, (long long)L_SEQ * HIDDEN,
        WinB + (size_t)D_INNER * HIDDEN, HIDDEN,
        zlast, D_INNER, B_SZ, D_INNER, HIDDEN);

    // 4) depthwise conv + SiLU
    conv_silu_kernel<<<cdiv((long long)BL * D_INNER, 256), blk, 0, stream>>>(xinRaw, conv_W, conv_b, xinSilu);

    // 5) GEMM2: x_dbl = xin_silu @ x_proj_W^T (M=8208,N=96,K=2048, f32 out)
    wmma_gemm_kernel<0><<<dim3(cdiv(BL, 64), 1), blk, 0, stream>>>(
        xinSilu, D_INNER, WxpB, D_INNER, xdbl, DT_RANK + 2 * D_STATE,
        BL, DT_RANK + 2 * D_STATE, D_INNER);

    // 6) dt-rank slice -> bf16, then GEMM3: dtraw = dt @ dt_proj_W^T (M=8208,N=2048,K=64)
    extract_dt_kernel<<<cdiv((long long)BL * DT_RANK, 256), blk, 0, stream>>>(xdbl, dtB);
    wmma_gemm_kernel<0><<<dim3(cdiv(BL, 64), D_INNER / 256), blk, 0, stream>>>(
        dtB, DT_RANK, WdtB, DT_RANK, dtraw, D_INNER, BL, D_INNER, DT_RANK);

    // 7) selective scan (513 sequential steps, 32768 parallel channels)
    scan_kernel<<<dim3(B_SZ * (D_INNER / 256)), blk, 0, stream>>>(
        dtraw, dt_proj_b, xinSilu, xdbl, A_log, D_param, zlast, ylast, ylastB);

    // 8) GEMM4: current = y_last @ out_proj_W^T (M=16,N=1024,K=2048) -> d_out current slot
    wmma_gemm_kernel<0><<<dim3(1, cdiv(HIDDEN, 256)), blk, 0, stream>>>(
        ylastB, D_INNER, WoutB, D_INNER, current, HIDDEN, B_SZ, HIDDEN, D_INNER);

    // 9) heads + memory shift
    heads_kernel<<<dim3(1), dim3(320), 0, stream>>>(current, actor_W, actor_b, critic_W, critic_b, logits, value);
    shift_mem_kernel<<<cdiv((long long)B_SZ * MEMLEN * HIDDEN, 256), blk, 0, stream>>>(mem, current, newmem);
}
